// Attention_NodeLevel_12378095747833
// MI455X (gfx1250) — matrix operand order repaired and checked by reading, not compiled, not run on hardware
//
#include <hip/hip_runtime.h>
#include <hip/hip_bf16.h>

// ---------------------------------------------------------------------------
// Fused graph-attention (GAT-style) for MI455X / gfx1250, wave32 + WMMA f16.
// Single pass over adj (flash-style online softmax), transposed-GEMM layout:
//   outT = x2T(16x32 A-tiles, per-wave) x [pT | adjT](32x16 B-frags, shared).
// A-operand tiles staged by the Tensor Data Mover (tensor_load_to_lds) from a
// pre-transposed f16 copy of input2, double-buffered in LDS (TENSORcnt sync).
// ---------------------------------------------------------------------------

#define NROWS 8192
#define MCOLS 8192
#define DDIM  256

typedef __attribute__((ext_vector_type(16))) _Float16     v16h;
typedef __attribute__((ext_vector_type(8)))  float        v8f;
typedef __attribute__((ext_vector_type(4)))  float        f4v;
typedef __attribute__((ext_vector_type(4)))  unsigned int v4u;
typedef __attribute__((ext_vector_type(8)))  int          v8i;
typedef __attribute__((ext_vector_type(4)))  int          v4i;

union V16H { v16h h; uint4 q[2]; };
union V8F  { v8f  f; f4v   q[2]; };

// ---------------------------------------------------------------------------
// Kernel 0: x2T[d][k] = (f16) input2[k][d]  -- one-time transpose+convert.
// ---------------------------------------------------------------------------
__global__ __launch_bounds__(256) void gat_x2t_kernel(
    const float* __restrict__ x2, _Float16* __restrict__ x2t)
{
    const int tid = threadIdx.x;              // owns column d = tid
    const int k0  = blockIdx.x * 32;
    __align__(16) _Float16 tmp[32];
#pragma unroll
    for (int k = 0; k < 32; ++k)              // coalesced 1KB row sweeps
        tmp[k] = (_Float16)x2[(size_t)(k0 + k) * DDIM + tid];
    uint4* dst = (uint4*)&x2t[(size_t)tid * MCOLS + k0];
    const uint4* src = (const uint4*)tmp;
#pragma unroll
    for (int q = 0; q < 4; ++q) dst[q] = src[q];
}

// ---------------------------------------------------------------------------
// Kernel 1: e1 = input1 @ a1  (N),  e2 = input2 @ a2  (M). One wave per row.
// ---------------------------------------------------------------------------
__global__ __launch_bounds__(256) void gat_ev_kernel(
    const float* __restrict__ x1, const float* __restrict__ x2,
    const float* __restrict__ a1, const float* __restrict__ a2,
    float* __restrict__ e1, float* __restrict__ e2)
{
    const int wid  = (blockIdx.x * blockDim.x + threadIdx.x) >> 5;
    const int lane = threadIdx.x & 31;

    const float* x; const float* a; float* o; int row;
    if (wid < NROWS) { x = x1; a = a1; o = e1; row = wid; }
    else             { x = x2; a = a2; o = e2; row = wid - NROWS; }

    float acc = 0.f;
#pragma unroll
    for (int i = 0; i < DDIM / 32; ++i) {
        const int c = lane + i * 32;
        acc += x[(size_t)row * DDIM + c] * a[c];
    }
#pragma unroll
    for (int off = 16; off >= 1; off >>= 1) acc += __shfl_xor(acc, off, 32);
    if (lane == 0) o[row] = acc;
}

// ---------------------------------------------------------------------------
// TDM: load 2D tile x2T[0:256][k0:k0+32] (f16) into LDS at lds_byte_addr.
// Tile row = 32 elem * 2B = 64B; pad 4 DWORDs per 16 DWORDs stored ->
// LDS row stride 80B (= the 40-half padded layout the A-frag reads expect).
// D# packing per CDNA5 ISA 8.3/8.4 (group0 128b, group1 256b).
// This toolchain exposes the 6-arg builtin:
//   (uint32x4 g0, int32x8 g1, int32x4 g2, int32x4 g3, int32x8 zpad, i32 cpol)
// ---------------------------------------------------------------------------
__device__ __forceinline__ void tdm_load_x2t_tile(const _Float16* gsrc,
                                                  unsigned int lds_byte_addr)
{
    const unsigned long long ga = (unsigned long long)(uintptr_t)gsrc;
    v4u g0 = { 1u,                                     // count=1 (valid D#)
               lds_byte_addr,                          // lds_addr
               (unsigned int)(ga & 0xFFFFFFFFu),       // global_addr[31:0]
               (unsigned int)((ga >> 32) & 0x01FFFFFFu) | (2u << 30) }; // [56:32]|type=2
    v8i g1;
    g1[0] = (1 << 16)      // data_size = 1 -> 2-byte elements
          | (1 << 20)      // pad_enable
          | (3 << 22)      // pad_interval: 2^(3+1) DWORDs = 64B of data
          | (3 << 25);     // pad_amount: 3+1 = 4 DWORDs (16B) of pad
    g1[1] = (int)((MCOLS & 0xFFFFu) << 16);            // tensor_dim0[15:0]
    g1[2] = (int)((DDIM  & 0xFFFFu) << 16);            // dim0 hi=0 | tensor_dim1[15:0]
    g1[3] = (int)(32u << 16);                          // dim1 hi=0 | tile_dim0=32
    g1[4] = DDIM;                                      // tile_dim1=256, tile_dim2=0
    g1[5] = MCOLS;                                     // tensor_dim0_stride[31:0]
    g1[6] = 0;                                         // stride hi | dim1_stride lo
    g1[7] = 0;
    v4i g2   = { 0, 0, 0, 0 };
    v4i g3   = { 0, 0, 0, 0 };
    v8i zpad = { 0, 0, 0, 0, 0, 0, 0, 0 };
    __builtin_amdgcn_tensor_load_to_lds(g0, g1, g2, g3, zpad, 0);
}

// ---------------------------------------------------------------------------
// Kernel 2a (TDM variant): fused masked-softmax + degree/gamma blend + SpMM.
// 8 waves per workgroup, 16 output rows per workgroup, wave w owns the
// 32-wide D-slice [32w, 32w+32). Double-buffered TDM staging of x2T tiles.
// ---------------------------------------------------------------------------
__global__ __launch_bounds__(256) void gat_main_tdm_kernel(
    const _Float16* __restrict__ x2t, const float* __restrict__ adj,
    const float* __restrict__ e1g,    const float* __restrict__ e2g,
    float* __restrict__ out)
{
    __shared__ __align__(16) _Float16 xbuf[2][DDIM][40]; // 2 x 20KB TDM tiles
    __shared__ __align__(16) _Float16 pfrag[512];        // B-frag: softmax numerator
    __shared__ __align__(16) _Float16 afrag[512];        // B-frag: raw adjacency
    __shared__ float mS[16], sS[16], degS[16], scaleS[16], e1s[16];

    const int tid  = threadIdx.x;
    const int wave = tid >> 5;
    const int lane = tid & 31;
    const int row0 = blockIdx.x * 16;

    const int r     = lane & 15;          // output row (A-row / B-col / state idx)
    const int kbase = lane & 16;          // producer & B-frag K range: 0 or 16
    const int kb    = (lane >> 4) << 3;   // A-side K half: kb>>3 picks K 0..15 or 16..31
    const int d0    = wave * 32;          // this wave's D-slice

    if (tid < 16) {
        e1s[tid]  = e1g[row0 + tid];
        mS[tid]   = -1e30f;
        sS[tid]   = 0.f;
        degS[tid] = 0.f;
    }

    // prologue: stage first tile into buffer 0
    if (wave == 0)
        tdm_load_x2t_tile(x2t, (unsigned int)(uintptr_t)&xbuf[0][0][0]);

    v8f accP0{}, accP1{};                 // exp-weighted numerator tiles (outT)
    v8f accB0{}, accB1{};                 // raw-adjacency GEMM tiles (outT)

    for (int k0 = 0; k0 < MCOLS; k0 += 32) {
        const int cur = (k0 >> 5) & 1;
        __syncthreads();                  // both LDS buffers free of readers

        // ---- wave 0: kick next TDM tile, then fence current one ----
        if (wave == 0) {
            if (k0 + 32 < MCOLS) {
                tdm_load_x2t_tile(x2t + (k0 + 32),
                                  (unsigned int)(uintptr_t)&xbuf[1 - cur][0][0]);
                __builtin_amdgcn_s_wait_tensorcnt(1);  // current tile complete
            } else {
                __builtin_amdgcn_s_wait_tensorcnt(0);
            }
        }

        // ---- rotating producer wave: masked scores + online softmax ----
        if (wave == ((k0 >> 5) & 7)) {
            const float e1v = e1s[r];
            const float* arow = &adj[(size_t)(row0 + r) * MCOLS + k0 + kbase];
            f4v a4[4], e2v[4];
#pragma unroll
            for (int q = 0; q < 4; ++q)
                a4[q] = __builtin_nontemporal_load(((const f4v*)arow) + q);
#pragma unroll
            for (int q = 0; q < 4; ++q)
                e2v[q] = ((const f4v*)(e2g + k0 + kbase))[q];
            if (k0 + 256 < MCOLS)          // this wave's next production chunk
                __builtin_prefetch(arow + 256, 0, 0);

            float av[16], ev[16];
            float pmax = -1e30f, dsum = 0.f;
#pragma unroll
            for (int t = 0; t < 16; ++t) {
                const float a = ((const float*)a4)[t];
                float e = e1v + ((const float*)e2v)[t];
                e = (e > 0.f) ? e : 0.2f * e;          // LeakyReLU(0.2)
                av[t] = a; ev[t] = e;
                pmax  = (a > 0.f) ? fmaxf(pmax, e) : pmax;
                dsum += a;
            }
            pmax = fmaxf(pmax, __shfl_xor(pmax, 16, 32));
            dsum += __shfl_xor(dsum, 16, 32);

            const float mold = mS[r];
            const float mnew = fmaxf(mold, pmax);
            const float scl  = __expf(mold - mnew);    // both -1e30 -> 1.0

            v16h ph{}, ah{};
            float psum = 0.f;
#pragma unroll
            for (int t = 0; t < 16; ++t) {
                const float p = (av[t] > 0.f) ? __expf(ev[t] - mnew) : 0.f;
                psum += p;
                ph[t] = (_Float16)p;
                ah[t] = (_Float16)((av[t] > 0.f) ? 1.f : 0.f);
            }
            psum += __shfl_xor(psum, 16, 32);

            if (lane < 16) {
                mS[r]     = mnew;
                sS[r]     = sS[r] * scl + psum;
                degS[r]   = degS[r] + dsum;
                scaleS[r] = scl;
            }
            { V16H u; u.h = ph; uint4* d = (uint4*)&pfrag[lane * 16]; d[0] = u.q[0]; d[1] = u.q[1]; }
            { V16H u; u.h = ah; uint4* d = (uint4*)&afrag[lane * 16]; d[0] = u.q[0]; d[1] = u.q[1]; }
        }
        __syncthreads();                  // frags + TDM tile visible

        // ---- per-lane rescale (C column = output row = lane&15) ----
        const float scl = scaleS[r];
#pragma unroll
        for (int i = 0; i < 8; ++i) { accP0[i] *= scl; accP1[i] *= scl; }

        // ---- shared B fragments: lane-contiguous b128 loads ----
        v16h bp, ba;
        { V16H u; const uint4* s = (const uint4*)&pfrag[lane * 16]; u.q[0] = s[0]; u.q[1] = s[1]; bp = u.h; }
        { V16H u; const uint4* s = (const uint4*)&afrag[lane * 16]; u.q[0] = s[0]; u.q[1] = s[1]; ba = u.h; }

        // ---- per-wave A fragments from the TDM-staged tile ----
        const int qi = kb >> 3;           // 0 or 1
        const _Float16* xb = &xbuf[cur][0][0];
        v16h a0, a1;
        { V16H u; const uint4* s = (const uint4*)&xb[(d0 + r) * 40];
          u.q[0] = s[2 * qi]; u.q[1] = s[2 * qi + 1]; a0 = u.h; }
        { V16H u; const uint4* s = (const uint4*)&xb[(d0 + 16 + r) * 40];
          u.q[0] = s[2 * qi]; u.q[1] = s[2 * qi + 1]; a1 = u.h; }

        // ---- four WMMA f32.16x16x32.f16 (EXEC all-ones here) ----
        accP0 = __builtin_amdgcn_wmma_f32_16x16x32_f16(false, a0, false, bp,
                                                       (short)0, accP0, false, false);
        accB0 = __builtin_amdgcn_wmma_f32_16x16x32_f16(false, a0, false, ba,
                                                       (short)0, accB0, false, false);
        accP1 = __builtin_amdgcn_wmma_f32_16x16x32_f16(false, a1, false, bp,
                                                       (short)0, accP1, false, false);
        accB1 = __builtin_amdgcn_wmma_f32_16x16x32_f16(false, a1, false, ba,
                                                       (short)0, accB1, false, false);
    }

    // ---- epilogue: out = (gamma*deg/s) * accP + (1-gamma) * accB ----
    const float sv   = sS[r];
    const float coef = (sv > 0.f) ? (0.1f * degS[r] / sv) : 0.f;
    const int   irow = row0 + r;
    const int   dlo  = d0 + ((lane < 16) ? 0 : 8);

    V8F o;
#pragma unroll
    for (int v = 0; v < 8; ++v) o.f[v] = coef * accP0[v] + 0.9f * accB0[v];
    *(f4v*)&out[(size_t)irow * DDIM + dlo]      = o.q[0];
    *(f4v*)&out[(size_t)irow * DDIM + dlo + 4]  = o.q[1];
#pragma unroll
    for (int v = 0; v < 8; ++v) o.f[v] = coef * accP1[v] + 0.9f * accB1[v];
    *(f4v*)&out[(size_t)irow * DDIM + dlo + 16] = o.q[0];
    *(f4v*)&out[(size_t)irow * DDIM + dlo + 20] = o.q[1];
}

// ---------------------------------------------------------------------------
// Kernel 2b (fallback, no TDM workspace): same algorithm, x2 staged by the
// threads (f32 loads + cvt + transposed LDS stores) each chunk.
// ---------------------------------------------------------------------------
__global__ __launch_bounds__(256) void gat_main_kernel(
    const float* __restrict__ x2,  const float* __restrict__ adj,
    const float* __restrict__ e1g, const float* __restrict__ e2g,
    float* __restrict__ out)
{
    __shared__ __align__(16) _Float16 xT[DDIM][40];
    __shared__ __align__(16) _Float16 pfrag[512];
    __shared__ __align__(16) _Float16 afrag[512];
    __shared__ float mS[16], sS[16], degS[16], scaleS[16], e1s[16];

    const int tid  = threadIdx.x;
    const int wave = tid >> 5;
    const int lane = tid & 31;
    const int row0 = blockIdx.x * 16;

    const int r     = lane & 15;
    const int kbase = lane & 16;
    const int kb    = (lane >> 4) << 3;
    const int d0    = wave * 32;

    if (tid < 16) {
        e1s[tid]  = e1g[row0 + tid];
        mS[tid]   = -1e30f;
        sS[tid]   = 0.f;
        degS[tid] = 0.f;
    }

    v8f accP0{}, accP1{}, accB0{}, accB1{};

    for (int k0 = 0; k0 < MCOLS; k0 += 32) {
        __syncthreads();

        {
            __align__(16) _Float16 tmp[32];
#pragma unroll
            for (int k = 0; k < 32; ++k)
                tmp[k] = (_Float16)x2[(size_t)(k0 + k) * DDIM + tid];
            uint4* dst = (uint4*)&xT[tid][0];
            const uint4* src = (const uint4*)tmp;
#pragma unroll
            for (int q = 0; q < 4; ++q) dst[q] = src[q];
        }

        if (wave == ((k0 >> 5) & 7)) {
            const float e1v = e1s[r];
            const float* arow = &adj[(size_t)(row0 + r) * MCOLS + k0 + kbase];
            f4v a4[4], e2v[4];
#pragma unroll
            for (int q = 0; q < 4; ++q)
                a4[q] = __builtin_nontemporal_load(((const f4v*)arow) + q);
#pragma unroll
            for (int q = 0; q < 4; ++q)
                e2v[q] = ((const f4v*)(e2g + k0 + kbase))[q];
            if (k0 + 256 < MCOLS)
                __builtin_prefetch(arow + 256, 0, 0);

            float av[16], ev[16];
            float pmax = -1e30f, dsum = 0.f;
#pragma unroll
            for (int t = 0; t < 16; ++t) {
                const float a = ((const float*)a4)[t];
                float e = e1v + ((const float*)e2v)[t];
                e = (e > 0.f) ? e : 0.2f * e;
                av[t] = a; ev[t] = e;
                pmax  = (a > 0.f) ? fmaxf(pmax, e) : pmax;
                dsum += a;
            }
            pmax = fmaxf(pmax, __shfl_xor(pmax, 16, 32));
            dsum += __shfl_xor(dsum, 16, 32);

            const float mold = mS[r];
            const float mnew = fmaxf(mold, pmax);
            const float scl  = __expf(mold - mnew);

            v16h ph{}, ah{};
            float psum = 0.f;
#pragma unroll
            for (int t = 0; t < 16; ++t) {
                const float p = (av[t] > 0.f) ? __expf(ev[t] - mnew) : 0.f;
                psum += p;
                ph[t] = (_Float16)p;
                ah[t] = (_Float16)((av[t] > 0.f) ? 1.f : 0.f);
            }
            psum += __shfl_xor(psum, 16, 32);

            if (lane < 16) {
                mS[r]     = mnew;
                sS[r]     = sS[r] * scl + psum;
                degS[r]   = degS[r] + dsum;
                scaleS[r] = scl;
            }
            { V16H u; u.h = ph; uint4* d = (uint4*)&pfrag[lane * 16]; d[0] = u.q[0]; d[1] = u.q[1]; }
            { V16H u; u.h = ah; uint4* d = (uint4*)&afrag[lane * 16]; d[0] = u.q[0]; d[1] = u.q[1]; }
        }
        __syncthreads();

        const float scl = scaleS[r];
#pragma unroll
        for (int i = 0; i < 8; ++i) { accP0[i] *= scl; accP1[i] *= scl; }

        v16h bp, ba;
        { V16H u; const uint4* s = (const uint4*)&pfrag[lane * 16]; u.q[0] = s[0]; u.q[1] = s[1]; bp = u.h; }
        { V16H u; const uint4* s = (const uint4*)&afrag[lane * 16]; u.q[0] = s[0]; u.q[1] = s[1]; ba = u.h; }

        const int qi = kb >> 3;
        v16h a0, a1;
        { V16H u; const uint4* s = (const uint4*)&xT[d0 + r][0];
          u.q[0] = s[2 * qi]; u.q[1] = s[2 * qi + 1]; a0 = u.h; }
        { V16H u; const uint4* s = (const uint4*)&xT[d0 + 16 + r][0];
          u.q[0] = s[2 * qi]; u.q[1] = s[2 * qi + 1]; a1 = u.h; }

        accP0 = __builtin_amdgcn_wmma_f32_16x16x32_f16(false, a0, false, bp,
                                                       (short)0, accP0, false, false);
        accB0 = __builtin_amdgcn_wmma_f32_16x16x32_f16(false, a0, false, ba,
                                                       (short)0, accB0, false, false);
        accP1 = __builtin_amdgcn_wmma_f32_16x16x32_f16(false, a1, false, bp,
                                                       (short)0, accP1, false, false);
        accB1 = __builtin_amdgcn_wmma_f32_16x16x32_f16(false, a1, false, ba,
                                                       (short)0, accB1, false, false);
    }

    const float sv   = sS[r];
    const float coef = (sv > 0.f) ? (0.1f * degS[r] / sv) : 0.f;
    const int   irow = row0 + r;
    const int   dlo  = d0 + ((lane < 16) ? 0 : 8);

    V8F o;
#pragma unroll
    for (int v = 0; v < 8; ++v) o.f[v] = coef * accP0[v] + 0.9f * accB0[v];
    *(f4v*)&out[(size_t)irow * DDIM + dlo]      = o.q[0];
    *(f4v*)&out[(size_t)irow * DDIM + dlo + 4]  = o.q[1];
#pragma unroll
    for (int v = 0; v < 8; ++v) o.f[v] = coef * accP1[v] + 0.9f * accB1[v];
    *(f4v*)&out[(size_t)irow * DDIM + dlo + 16] = o.q[0];
    *(f4v*)&out[(size_t)irow * DDIM + dlo + 20] = o.q[1];
}

// ---------------------------------------------------------------------------
extern "C" void kernel_launch(void* const* d_in, const int* in_sizes, int n_in,
                              void* d_out, int out_size, void* d_ws, size_t ws_size,
                              hipStream_t stream) {
    const float* input1 = (const float*)d_in[0];
    const float* input2 = (const float*)d_in[1];
    const float* adj    = (const float*)d_in[2];
    const float* a1     = (const float*)d_in[3];
    const float* a2     = (const float*)d_in[4];
    float* out = (float*)d_out;

    float* e1 = (float*)d_ws;                       // N floats
    float* e2 = e1 + NROWS;                         // M floats
    const size_t ev_bytes  = (size_t)(NROWS + MCOLS) * sizeof(float);
    const size_t tdm_bytes = ev_bytes + (size_t)DDIM * MCOLS * sizeof(_Float16);

    gat_ev_kernel<<<(NROWS + MCOLS) / 8, 256, 0, stream>>>(input1, input2, a1, a2, e1, e2);

    if (ws_size >= tdm_bytes) {
        _Float16* x2t = (_Float16*)((char*)d_ws + ev_bytes);   // 16B-aligned
        gat_x2t_kernel<<<MCOLS / 32, 256, 0, stream>>>(input2, x2t);
        gat_main_tdm_kernel<<<NROWS / 16, 256, 0, stream>>>(x2t, adj, e1, e2, out);
    } else {
        gat_main_kernel<<<NROWS / 16, 256, 0, stream>>>(input2, adj, e1, e2, out);
    }
}
